// DecoderC_85925115724578
// MI455X (gfx1250) — matrix operand order repaired and checked by reading, not compiled, not run on hardware
//
#include <hip/hip_runtime.h>
#include <math.h>

typedef __attribute__((ext_vector_type(16))) __bf16 v16bf;
typedef __attribute__((ext_vector_type(8)))  float  v8f;

#define EPSF 1e-5f

__device__ __forceinline__ float gelu_exact(float x) {
  return 0.5f * x * (1.0f + erff(x * 0.70710678118654752440f));
}
__device__ __forceinline__ float sigmoidf_(float x) {
  return 1.0f / (1.0f + __expf(-x));
}

// Unconditional: 8 consecutive fp32 -> bf16 fragment elements [base, base+8).
__device__ __forceinline__ void ld8u(v16bf& f, int base, const float* __restrict__ p) {
  float4 x0 = *(const float4*)p;
  float4 x1 = *(const float4*)(p + 4);
  f[base + 0] = (__bf16)x0.x; f[base + 1] = (__bf16)x0.y;
  f[base + 2] = (__bf16)x0.z; f[base + 3] = (__bf16)x0.w;
  f[base + 4] = (__bf16)x1.x; f[base + 5] = (__bf16)x1.y;
  f[base + 6] = (__bf16)x1.z; f[base + 7] = (__bf16)x1.w;
}

// Full-chunk B fragment (no bounds checks).  transB=1 is contiguous per lane
// (4 x b128); transB=0 is 16 branchless strided b32.
__device__ __forceinline__ v16bf loadB_full(const float* __restrict__ Bm, int ldb,
                                            int transB, int bcol, int kb) {
  v16bf fb;
  if (transB) {
    const float* p = Bm + (size_t)bcol * ldb + kb;
    ld8u(fb, 0, p);
    ld8u(fb, 8, p + 8);
  } else {
    const float* p = Bm + (size_t)kb * ldb + bcol;
#pragma unroll
    for (int e = 0; e < 16; ++e) fb[e] = (__bf16)p[(size_t)e * ldb];
  }
  return fb;
}

// Tail B fragment for the 16-wide K remainder (K%32==16).  All lanes load the
// valid k-range [Kb, Kb+16); upper-half lanes (logical k >= K) zero their
// fragment with a value select (v_cndmask) -- no exec divergence.
__device__ __forceinline__ v16bf loadB_tail(const float* __restrict__ Bm, int ldb,
                                            int transB, int bcol, int Kb, bool z) {
  v16bf fb;
  if (transB) {
    const float* p = Bm + (size_t)bcol * ldb + Kb;
    float4 x0 = *(const float4*)p;
    float4 x1 = *(const float4*)(p + 4);
    float4 x2 = *(const float4*)(p + 8);
    float4 x3 = *(const float4*)(p + 12);
    fb[0]  = (__bf16)(z ? 0.f : x0.x); fb[1]  = (__bf16)(z ? 0.f : x0.y);
    fb[2]  = (__bf16)(z ? 0.f : x0.z); fb[3]  = (__bf16)(z ? 0.f : x0.w);
    fb[4]  = (__bf16)(z ? 0.f : x1.x); fb[5]  = (__bf16)(z ? 0.f : x1.y);
    fb[6]  = (__bf16)(z ? 0.f : x1.z); fb[7]  = (__bf16)(z ? 0.f : x1.w);
    fb[8]  = (__bf16)(z ? 0.f : x2.x); fb[9]  = (__bf16)(z ? 0.f : x2.y);
    fb[10] = (__bf16)(z ? 0.f : x2.z); fb[11] = (__bf16)(z ? 0.f : x2.w);
    fb[12] = (__bf16)(z ? 0.f : x3.x); fb[13] = (__bf16)(z ? 0.f : x3.y);
    fb[14] = (__bf16)(z ? 0.f : x3.z); fb[15] = (__bf16)(z ? 0.f : x3.w);
  } else {
    const float* p = Bm + (size_t)Kb * ldb + bcol;
#pragma unroll
    for (int e = 0; e < 16; ++e) {
      float bv = p[(size_t)e * ldb];
      fb[e] = (__bf16)(z ? 0.f : bv);
    }
  }
  return fb;
}

// ---------------------------------------------------------------------------
// WMMA GEMM:  Y[M,N] = act( alpha * (A[M,K] @ B) + bias[N] + Cadd[M,N] )
// REQUIRES M%16==0, N%16==0, K%16==0 (true for every launch here).
// One wave computes a 16x64 strip: 4 N-subtiles sharing one A fragment,
// 4 v_wmma_f32_16x16x32_bf16 per 32-K chunk, fp32 accumulate.  Branch-free
// inner loop; single uniform tail when K%32==16.
// ---------------------------------------------------------------------------
__global__ void __launch_bounds__(128)
k_gemm(const float* __restrict__ A, int lda,
       const float* __restrict__ Bm, int ldb, int transB,
       const float* __restrict__ bias,
       const float* __restrict__ Cadd, int ldc,
       float* __restrict__ Y, int ldy,
       int M, int N, int K, float alpha, int act)
{
  const int tilesN  = N >> 4;
  const int tilesM  = M >> 4;
  const int tilesNS = (tilesN + 3) >> 2;          // supertiles of 4 N-tiles
  int st = blockIdx.x * 4 + (threadIdx.x >> 5);
  if (st >= tilesM * tilesNS) return;
  const int tM  = st / tilesNS;
  const int tN0 = (st - tM * tilesNS) * 4;        // first N-tile of strip
  const int nt  = (tilesN - tN0 < 4) ? (tilesN - tN0) : 4;

  const int lane = threadIdx.x & 31;
  const int half = lane >> 4;
  const int lr   = lane & 15;
  const int arow = tM * 16 + lr;

  const int bc0 = (tN0 + 0) * 16 + lr;
  const int bc1 = (tN0 + 1) * 16 + lr;
  const int bc2 = (tN0 + 2) * 16 + lr;
  const int bc3 = (tN0 + 3) * 16 + lr;

  v8f acc0 = {0.f,0.f,0.f,0.f,0.f,0.f,0.f,0.f};
  v8f acc1 = acc0, acc2 = acc0, acc3 = acc0;

  const float* Arow = A + (size_t)arow * lda;
  const int Kfull = K & ~31;

  for (int k0 = 0; k0 < Kfull; k0 += 32) {
    v16bf fa;   // lanes0-15: K 0..15; lanes16-31: K 16..31
    ld8u(fa, 0, Arow + k0 + half * 16);
    ld8u(fa, 8, Arow + k0 + half * 16 + 8);
    const int kb = k0 + half * 16;   // B: lanes0-15 K 0..15; lanes16-31 K 16..31
    {
      v16bf fb = loadB_full(Bm, ldb, transB, bc0, kb);
      acc0 = __builtin_amdgcn_wmma_f32_16x16x32_bf16(false, fa, false, fb, (short)0, acc0, false, false);
    }
    if (nt > 1) {
      v16bf fb = loadB_full(Bm, ldb, transB, bc1, kb);
      acc1 = __builtin_amdgcn_wmma_f32_16x16x32_bf16(false, fa, false, fb, (short)0, acc1, false, false);
    }
    if (nt > 2) {
      v16bf fb = loadB_full(Bm, ldb, transB, bc2, kb);
      acc2 = __builtin_amdgcn_wmma_f32_16x16x32_bf16(false, fa, false, fb, (short)0, acc2, false, false);
    }
    if (nt > 3) {
      v16bf fb = loadB_full(Bm, ldb, transB, bc3, kb);
      acc3 = __builtin_amdgcn_wmma_f32_16x16x32_bf16(false, fa, false, fb, (short)0, acc3, false, false);
    }
  }

  if (K & 16) {                      // uniform tail: logical k in [Kfull, K)
    const int Kb = Kfull;
    const bool z = (half != 0);      // upper-half lanes' fragment is all-zero k
    v16bf fa;
    ld8u(fa, 0, Arow + Kb);   // always in bounds (16 valid k's)
    ld8u(fa, 8, Arow + Kb + 8);
    {
      v16bf fb = loadB_tail(Bm, ldb, transB, bc0, Kb, z);
      acc0 = __builtin_amdgcn_wmma_f32_16x16x32_bf16(false, fa, false, fb, (short)0, acc0, false, false);
    }
    if (nt > 1) {
      v16bf fb = loadB_tail(Bm, ldb, transB, bc1, Kb, z);
      acc1 = __builtin_amdgcn_wmma_f32_16x16x32_bf16(false, fa, false, fb, (short)0, acc1, false, false);
    }
    if (nt > 2) {
      v16bf fb = loadB_tail(Bm, ldb, transB, bc2, Kb, z);
      acc2 = __builtin_amdgcn_wmma_f32_16x16x32_bf16(false, fa, false, fb, (short)0, acc2, false, false);
    }
    if (nt > 3) {
      v16bf fb = loadB_tail(Bm, ldb, transB, bc3, Kb, z);
      acc3 = __builtin_amdgcn_wmma_f32_16x16x32_bf16(false, fa, false, fb, (short)0, acc3, false, false);
    }
  }

  const int row0 = tM * 16 + half * 8;
  auto epi = [&](const v8f& acc, int col) {
    float bsv = bias ? bias[col] : 0.f;
#pragma unroll
    for (int e = 0; e < 8; ++e) {
      int row = row0 + e;
      float v = alpha * acc[e] + bsv;
      if (Cadd) v += Cadd[(size_t)row * ldc + col];
      if (act == 1) v = gelu_exact(v);
      else if (act == 2) v = sigmoidf_(v);
      Y[(size_t)row * ldy + col] = v;
    }
  };
  epi(acc0, bc0);
  if (nt > 1) epi(acc1, bc1);
  if (nt > 2) epi(acc2, bc2);
  if (nt > 3) epi(acc3, bc3);
}

// ---------------------------------------------------------------------------
// LayerNorm over last dim C.  One wave per row.
// transHW==0: x contiguous [rows,C].  transHW=HW: x planar [B,C,HW],
// row = b*HW + hw.  Output always contiguous [rows,C].
// ---------------------------------------------------------------------------
__global__ void k_layernorm(const float* __restrict__ x,
                            const float* __restrict__ g,
                            const float* __restrict__ bt,
                            float* __restrict__ y, int C, int transHW)
{
  int row = blockIdx.x;
  int lane = threadIdx.x;
  float vbuf[12];
  int cnt = 0;
  float s = 0.f;
  if (transHW) {
    int hw = row % transHW;
    int bb = row / transHW;
    for (int c = lane; c < C; c += 32) {
      float v = x[((size_t)bb * C + c) * transHW + hw];
      vbuf[cnt++] = v; s += v;
    }
  } else {
    for (int c = lane; c < C; c += 32) {
      float v = x[(size_t)row * C + c];
      vbuf[cnt++] = v; s += v;
    }
  }
  for (int off = 16; off; off >>= 1) s += __shfl_xor(s, off, 32);
  float mean = s / (float)C;
  float q = 0.f;
  for (int i = 0; i < cnt; ++i) { float d = vbuf[i] - mean; q += d * d; }
  for (int off = 16; off; off >>= 1) q += __shfl_xor(q, off, 32);
  float rstd = rsqrtf(q / (float)C + EPSF);
  cnt = 0;
  for (int c = lane; c < C; c += 32)
    y[(size_t)row * C + c] = (vbuf[cnt++] - mean) * rstd * g[c] + bt[c];
}

// ---------------------------------------------------------------------------
__global__ void k_softmax(float* __restrict__ s, int cols)
{
  __shared__ float red[256];
  float* row = s + (size_t)blockIdx.x * cols;
  int tid = threadIdx.x;
  float m = -3.4e38f;
  for (int c = tid; c < cols; c += 256) m = fmaxf(m, row[c]);
  red[tid] = m; __syncthreads();
  for (int off = 128; off; off >>= 1) {
    if (tid < off) red[tid] = fmaxf(red[tid], red[tid + off]);
    __syncthreads();
  }
  float mx = red[0]; __syncthreads();
  float sum = 0.f;
  for (int c = tid; c < cols; c += 256) {
    float e = __expf(row[c] - mx); row[c] = e; sum += e;
  }
  red[tid] = sum; __syncthreads();
  for (int off = 128; off; off >>= 1) {
    if (tid < off) red[tid] += red[tid + off];
    __syncthreads();
  }
  float inv = 1.f / red[0];
  for (int c = tid; c < cols; c += 256) row[c] *= inv;
}

// ---------------------------------------------------------------------------
// nn.Fold (gather form, race-free).  t = one batch of GEMM output
// [L, C*k*k] row-major; out = one batch planar [C, outsz, outsz].
// ---------------------------------------------------------------------------
__global__ void k_fold(const float* __restrict__ t, float* __restrict__ out,
                       int C, int ksz, int stride, int pad, int Lh, int Lw, int outsz)
{
  int idx = blockIdx.x * blockDim.x + threadIdx.x;
  int total = C * outsz * outsz;
  if (idx >= total) return;
  int xo = idx % outsz;
  int yo = (idx / outsz) % outsz;
  int c  = idx / (outsz * outsz);
  int y = yo + pad, x = xo + pad;
  int ckk = C * ksz * ksz;
  float s = 0.f;
  for (int ki = 0; ki < ksz; ++ki) {
    int ry = y - ki;
    if (ry < 0) break;
    if (ry % stride) continue;
    int lh = ry / stride;
    if (lh >= Lh) continue;
    for (int kj = 0; kj < ksz; ++kj) {
      int rx = x - kj;
      if (rx < 0) break;
      if (rx % stride) continue;
      int lw = rx / stride;
      if (lw >= Lw) continue;
      s += t[(size_t)(lh * Lw + lw) * ckk + c * ksz * ksz + ki * ksz + kj];
    }
  }
  out[idx] = s;
}

// ---------------------------------------------------------------------------
// Dense 3x3 conv, planar NCHW, pad=1, stride=1, + bias (+ residual = input).
// ---------------------------------------------------------------------------
__global__ void k_conv3x3(const float* __restrict__ in, const float* __restrict__ w,
                          const float* __restrict__ bias, float* __restrict__ out,
                          int Bn, int Cin, int Cout, int H, int W, int addres)
{
  int idx = blockIdx.x * blockDim.x + threadIdx.x;
  int total = Bn * Cout * H * W;
  if (idx >= total) return;
  int x = idx % W;
  int y = (idx / W) % H;
  int co = (idx / (W * H)) % Cout;
  int b = idx / (W * H * Cout);
  float s = bias[co];
  for (int ci = 0; ci < Cin; ++ci) {
    const float* ip = in + (size_t)(b * Cin + ci) * H * W;
    const float* wp = w + (size_t)(co * Cin + ci) * 9;
    for (int ky = 0; ky < 3; ++ky) {
      int iy = y + ky - 1;
      if (iy < 0 || iy >= H) continue;
      for (int kx = 0; kx < 3; ++kx) {
        int ix = x + kx - 1;
        if (ix < 0 || ix >= W) continue;
        s += wp[ky * 3 + kx] * ip[(size_t)iy * W + ix];
      }
    }
  }
  if (addres) s += in[((size_t)(b * Cin + co) * H + y) * W + x];
  out[idx] = s;
}

// ---------------------------------------------------------------------------
// Depthwise 3x3 conv (pad 1) on token-major [B, H*W, C] + bias + residual.
// ---------------------------------------------------------------------------
__global__ void k_dwconv(const float* __restrict__ in, const float* __restrict__ w,
                         const float* __restrict__ bias, float* __restrict__ out,
                         int Bn, int H, int W, int C)
{
  int idx = blockIdx.x * blockDim.x + threadIdx.x;
  int total = Bn * H * W * C;
  if (idx >= total) return;
  int c = idx % C;
  int hw = (idx / C) % (H * W);
  int b = idx / (C * H * W);
  int y = hw / W, x = hw % W;
  float s = bias[c];
  for (int ky = 0; ky < 3; ++ky) {
    int iy = y + ky - 1;
    if (iy < 0 || iy >= H) continue;
    for (int kx = 0; kx < 3; ++kx) {
      int ix = x + kx - 1;
      if (ix < 0 || ix >= W) continue;
      s += w[c * 9 + ky * 3 + kx] * in[((size_t)b * H * W + iy * W + ix) * C + c];
    }
  }
  out[idx] = s + in[idx];
}

// ---------------------------------------------------------------------------
// Depthwise sr x sr stride-sr conv + eval BatchNorm, token-major in/out.
// ---------------------------------------------------------------------------
__global__ void k_srconv_bn(const float* __restrict__ in, const float* __restrict__ w,
                            const float* __restrict__ wb,
                            const float* __restrict__ g, const float* __restrict__ bb,
                            const float* __restrict__ m, const float* __restrict__ v,
                            float* __restrict__ out, int Bn, int H, int W, int C, int sr)
{
  int OH = H / sr, OW = W / sr;
  int idx = blockIdx.x * blockDim.x + threadIdx.x;
  int total = Bn * OH * OW * C;
  if (idx >= total) return;
  int c = idx % C;
  int o = (idx / C) % (OH * OW);
  int b = idx / (C * OH * OW);
  int oy = o / OW, ox = o % OW;
  float s = wb[c];
  for (int ky = 0; ky < sr; ++ky)
    for (int kx = 0; kx < sr; ++kx)
      s += w[c * sr * sr + ky * sr + kx] *
           in[((size_t)b * H * W + (size_t)(oy * sr + ky) * W + (ox * sr + kx)) * C + c];
  s = (s - m[c]) * rsqrtf(v[c] + EPSF) * g[c] + bb[c];
  out[idx] = s;
}

// ---------------------------------------------------------------------------
__global__ void k_concat(const float* __restrict__ a, const float* __restrict__ b,
                         float* __restrict__ out, int rows, int C1, int C2)
{
  int C = C1 + C2;
  size_t idx = (size_t)blockIdx.x * blockDim.x + threadIdx.x;
  size_t total = (size_t)rows * C;
  if (idx >= total) return;
  int c = (int)(idx % C);
  size_t r = idx / C;
  out[idx] = (c < C1) ? a[r * C1 + c] : b[r * C2 + (c - C1)];
}

// Dot-product head (N=1) + sigmoid; one wave per row.
__global__ void k_head(const float* __restrict__ x, const float* __restrict__ w,
                       const float* __restrict__ bs, float* __restrict__ out, int K)
{
  int row = blockIdx.x;
  int lane = threadIdx.x;
  float s = 0.f;
  for (int c = lane; c < K; c += 32) s += x[(size_t)row * K + c] * w[c];
  for (int off = 16; off; off >>= 1) s += __shfl_xor(s, off, 32);
  if (lane == 0) out[row] = sigmoidf_(s + bs[0]);
}

// ===========================================================================
// Host orchestration
// ===========================================================================
struct Blk {
  const float *ao_b,*ao_w,*bn_b,*bn_g,*bn_m,*bn_v,*dw_b,*dw_w,
              *k_b,*k_w,*n1_b,*n1_g,*q_b,*q_w,*sr_b,*sr_w,*v_b,*v_w;
};
struct CAp { const float *k2,*proj_b,*proj_w,*q1,*v2; };
struct PEp {
  CAp ca;
  const float *cf_b1,*cf_b2,*cf_w1,*cf_w2,*lpu_b,*lpu_w,
              *norm2_b,*norm2_g,*norm3_b,*norm3_g,*norm_b,*norm_g,*proj_b,*proj_w;
};

extern "C" void kernel_launch(void* const* d_in, const int* in_sizes, int n_in,
                              void* d_out, int out_size, void* d_ws, size_t ws_size,
                              hipStream_t stream)
{
  (void)in_sizes; (void)n_in; (void)out_size; (void)ws_size;

  const float* fea16 = (const float*)d_in[0];   // [4,196,384]
  const float* fea8  = (const float*)d_in[1];   // [4,784,128]
  const float* fea4  = (const float*)d_in[2];   // [4,3136,64]

  // ---- unpack params (jax pytree flatten: dict keys sorted, lists in order)
  int pi = 3;
  auto nx = [&]() { return (const float*)d_in[pi++]; };
  Blk bc[6], bd[3];
  for (int i = 0; i < 6; ++i) {
    Blk& p = bc[i];
    p.ao_b=nx(); p.ao_w=nx(); p.bn_b=nx(); p.bn_g=nx(); p.bn_m=nx(); p.bn_v=nx();
    p.dw_b=nx(); p.dw_w=nx(); p.k_b=nx(); p.k_w=nx(); p.n1_b=nx(); p.n1_g=nx();
    p.q_b=nx(); p.q_w=nx(); p.sr_b=nx(); p.sr_w=nx(); p.v_b=nx(); p.v_w=nx();
  }
  for (int i = 0; i < 3; ++i) {
    Blk& p = bd[i];
    p.ao_b=nx(); p.ao_w=nx(); p.dw_b=nx(); p.dw_w=nx(); p.k_b=nx(); p.k_w=nx();
    p.n1_b=nx(); p.n1_g=nx(); p.q_b=nx(); p.q_w=nx(); p.v_b=nx(); p.v_w=nx();
    p.bn_b=p.bn_g=p.bn_m=p.bn_v=p.sr_b=p.sr_w=nullptr;
  }
  const float *mlp_b1=nx(), *mlp_b2=nx(), *mlp_w1=nx(), *mlp_w2=nx();
  const float *normP_b=nx(), *normP_g=nx();
  PEp pe_b{}, pe_c{}, pe_d{};
  pe_b.cf_b1=nx(); pe_b.cf_b2=nx(); pe_b.cf_w1=nx(); pe_b.cf_w2=nx();
  pe_b.lpu_b=nx(); pe_b.lpu_w=nx(); pe_b.norm2_b=nx(); pe_b.norm2_g=nx();
  pe_b.norm_b=nx(); pe_b.norm_g=nx(); pe_b.proj_b=nx(); pe_b.proj_w=nx();
  auto fill_pe = [&](PEp& p) {
    p.ca.k2=nx(); p.ca.proj_b=nx(); p.ca.proj_w=nx(); p.ca.q1=nx(); p.ca.v2=nx();
    p.cf_b1=nx(); p.cf_b2=nx(); p.cf_w1=nx(); p.cf_w2=nx();
    p.lpu_b=nx(); p.lpu_w=nx(); p.norm2_b=nx(); p.norm2_g=nx();
    p.norm3_b=nx(); p.norm3_g=nx(); p.norm_b=nx(); p.norm_g=nx();
    p.proj_b=nx(); p.proj_w=nx();
  };
  fill_pe(pe_c); fill_pe(pe_d);
  const float *pre16_b=nx(), *pre16_w=nx(), *pre1_b=nx(), *pre1_w=nx();
  const float *pre4_b=nx(),  *pre4_w=nx(),  *pre8_b=nx(), *pre8_w=nx();
  const float *rel_c=nx(), *rel_d=nx();

  // ---- workspace arena (floats)
  float* ws = (float*)d_ws;
  size_t off = 0;
  auto arena = [&](size_t n) { float* p = ws + off; off += n; return p; };
  float* W_T    = arena(3612672);    // max packed t (pe_c: 3136x1152)
  float* W_SC   = arena(9834496);    // 3136x3136 scores / per-batch pe_b t
  float* W_IMG1 = arena(12845056);   // planar img / tok (x0)
  float* W_IMG2 = arena(12845056);   // lpu out / hmid
  float* W_CAT  = arena(2408448);    // concat [12544,192]
  float* W_H    = arena(3211264);    // h / f1 [200704,16]
  float* W_X    = arena(802816);     // persistent block state
  float* W_Y    = arena(802816);
  float* W_Q    = arena(802816);
  float* W_K    = arena(802816);
  float* W_V    = arena(802816);
  float* W_KV   = arena(200704);
  float* W_O    = arena(802816);
  float* W_X2   = arena(802816);

  float* out_m16 = (float*)d_out;          // 784
  float* out_m8  = out_m16 + 784;          // 3136
  float* out_m4  = out_m8  + 3136;         // 12544
  float* out_m1  = out_m4  + 12544;        // 200704

  auto gemm = [&](const float* A, int lda, const float* Bm, int ldb, int transB,
                  const float* bias, const float* Cadd, int ldc,
                  float* Y, int ldy, int M, int N, int K, float alpha, int act) {
    int tilesN  = N / 16;
    int tilesNS = (tilesN + 3) / 4;
    int tiles   = (M / 16) * tilesNS;
    int blocks  = (tiles + 3) / 4;
    k_gemm<<<blocks, 128, 0, stream>>>(A, lda, Bm, ldb, transB, bias, Cadd, ldc,
                                       Y, ldy, M, N, K, alpha, act);
  };
  auto lnorm = [&](const float* x, const float* g, const float* bt,
                   float* y, int rows, int C, int transHW) {
    k_layernorm<<<rows, 32, 0, stream>>>(x, g, bt, y, C, transHW);
  };

  // ---- patch_embed -----------------------------------------------------
  auto run_pe = [&](const float* xtok, int L, int Hsp, int C, const float* enc, int E,
                    int ksz, int stride, int pad, const PEp& pe, bool fused, bool perBatch) {
    const int Nt = C * ksz * ksz;
    const int outsz = Hsp * stride;
    const int HW = outsz * outsz;
    if (!perBatch) {
      gemm(xtok, C, pe.proj_w, Nt, 0, pe.proj_b, nullptr, 0, W_T, Nt, 4 * L, Nt, C, 1.f, 0);
      for (int b = 0; b < 4; ++b)
        k_fold<<<(C * HW + 255) / 256, 256, 0, stream>>>(
            W_T + (size_t)b * L * Nt, W_IMG1 + (size_t)b * C * HW,
            C, ksz, stride, pad, Hsp, Hsp, outsz);
    } else {
      for (int b = 0; b < 4; ++b) {
        gemm(xtok + (size_t)b * L * C, C, pe.proj_w, Nt, 0, pe.proj_b, nullptr, 0,
             W_SC, Nt, L, Nt, C, 1.f, 0);
        k_fold<<<(C * HW + 255) / 256, 256, 0, stream>>>(
            W_SC, W_IMG1 + (size_t)b * C * HW, C, ksz, stride, pad, Hsp, Hsp, outsz);
      }
    }
    k_conv3x3<<<(4 * C * HW + 255) / 256, 256, 0, stream>>>(
        W_IMG1, pe.lpu_w, pe.lpu_b, W_IMG2, 4, C, C, outsz, outsz, 1);
    const int rows = 4 * HW;
    lnorm(W_IMG2, pe.norm_g, pe.norm_b, W_IMG1, rows, C, HW);   // tok = x0
    if (fused) {
      k_concat<<<(unsigned)(((size_t)rows * (C + E) + 255) / 256), 256, 0, stream>>>(
          W_IMG1, enc, W_CAT, rows, C, E);
      gemm(W_CAT, C + E, pe.cf_w1, C, 0, pe.cf_b1, nullptr, 0, W_IMG2, C, rows, C, C + E, 1.f, 1);
      gemm(W_IMG2, C, pe.cf_w2, E, 0, pe.cf_b2, nullptr, 0, W_H, E, rows, E, C, 1.f, 0);
      lnorm(W_H, pe.norm2_g, pe.norm2_b, W_H, rows, E, 0);
      // cross attention (heads=4)
      gemm(W_H,    E, pe.ca.q1, E, 0, nullptr, nullptr, 0, W_Q, E, rows, E, E, 1.f, 0);
      gemm(W_IMG1, C, pe.ca.k2, E, 0, nullptr, nullptr, 0, W_K, E, rows, E, C, 1.f, 0);
      gemm(W_IMG1, C, pe.ca.v2, E, 0, nullptr, nullptr, 0, W_V, E, rows, E, C, 1.f, 0);
      const int hd = E / 4, Nq = HW;
      const float alpha = 1.0f / sqrtf((float)hd);
      for (int b = 0; b < 4; ++b)
        for (int hh = 0; hh < 4; ++hh) {
          size_t o = (size_t)b * Nq * E + (size_t)hh * hd;
          gemm(W_Q + o, E, W_K + o, E, 1, nullptr, nullptr, 0, W_SC, Nq, Nq, Nq, hd, alpha, 0);
          k_softmax<<<Nq, 256, 0, stream>>>(W_SC, Nq);
          gemm(W_SC, Nq, W_V + o, E, 0, nullptr, nullptr, 0, W_O + o, E, Nq, hd, Nq, 1.f, 0);
        }
      gemm(W_O, E, pe.ca.proj_w, E, 0, pe.ca.proj_b, W_H, E, W_X, E, rows, E, E, 1.f, 0);
      lnorm(W_X, pe.norm3_g, pe.norm3_b, W_X, rows, E, 0);
    } else {
      gemm(W_IMG1, C, pe.cf_w1, C, 0, pe.cf_b1, nullptr, 0, W_IMG2, C, rows, C, C, 1.f, 1);
      gemm(W_IMG2, C, pe.cf_w2, E, 0, pe.cf_b2, nullptr, 0, W_H, E, rows, E, C, 1.f, 0);
      lnorm(W_H, pe.norm2_g, pe.norm2_b, W_H, rows, E, 0);
    }
  };

  // ---- transformer block ----------------------------------------------
  auto run_block = [&](const Blk& p, int D, int H, int W, int nh, int sr, const float* rel) {
    const int Ntok = H * W;
    const int rows = 4 * Ntok;
    lnorm(W_X, p.n1_g, p.n1_b, W_Y, rows, D, 0);
    gemm(W_Y, D, p.q_w, D, 0, p.q_b, nullptr, 0, W_Q, D, rows, D, D, 1.f, 0);
    const float* kvsrc = W_Y;
    int Nk = Ntok;
    if (sr > 1) {
      Nk = (H / sr) * (W / sr);
      k_srconv_bn<<<(4 * Nk * D + 255) / 256, 256, 0, stream>>>(
          W_Y, p.sr_w, p.sr_b, p.bn_g, p.bn_b, p.bn_m, p.bn_v, W_KV, 4, H, W, D, sr);
      kvsrc = W_KV;
    }
    gemm(kvsrc, D, p.k_w, D, 0, p.k_b, nullptr, 0, W_K, D, 4 * Nk, D, D, 1.f, 0);
    gemm(kvsrc, D, p.v_w, D, 0, p.v_b, nullptr, 0, W_V, D, 4 * Nk, D, D, 1.f, 0);
    const int hd = D / nh;
    const float alpha = 1.0f / sqrtf((float)hd);
    for (int b = 0; b < 4; ++b)
      for (int hh = 0; hh < nh; ++hh) {
        size_t qoff = (size_t)b * Ntok * D + (size_t)hh * hd;
        size_t koff = (size_t)b * Nk * D + (size_t)hh * hd;
        const float* radd = rel ? rel + (size_t)hh * Ntok * Nk : nullptr;
        gemm(W_Q + qoff, D, W_K + koff, D, 1, nullptr, radd, Nk, W_SC, Nk, Ntok, Nk, hd, alpha, 0);
        k_softmax<<<Ntok, 256, 0, stream>>>(W_SC, Nk);
        gemm(W_SC, Nk, W_V + koff, D, 0, nullptr, nullptr, 0, W_O + qoff, D, Ntok, hd, Nk, 1.f, 0);
      }
    gemm(W_O, D, p.ao_w, D, 0, p.ao_b, W_X, D, W_X2, D, rows, D, D, 1.f, 0);
    k_dwconv<<<(rows * D + 255) / 256, 256, 0, stream>>>(W_X2, p.dw_w, p.dw_b, W_X, 4, H, W, D);
  };

  // ======================= pipeline =====================================
  // Stage 1: 1/16 head
  lnorm(fea16, normP_g, normP_b, W_Y, 784, 384, 0);
  gemm(W_Y,  384, mlp_w1, 384, 0, mlp_b1, nullptr, 0, W_CAT, 384, 784, 384, 384, 1.f, 1);
  gemm(W_CAT,384, mlp_w2, 128, 0, mlp_b2, nullptr, 0, W_H,   128, 784, 128, 384, 1.f, 0);
  k_head<<<784, 32, 0, stream>>>(W_H, pre16_w, pre16_b, out_m16, 128);

  // Stage 2: pe_d (384ch 14x14 -> 128ch tokens @ 28x28) + 3 blocks_d
  run_pe(fea16, 196, 14, 384, fea8, 128, 3, 2, 1, pe_d, true, false);
  for (int i = 0; i < 3; ++i) run_block(bd[i], 128, 28, 28, 8, 1, rel_d);
  k_head<<<3136, 32, 0, stream>>>(W_X, pre8_w, pre8_b, out_m8, 128);

  // Stage 3: pe_c (128ch 28x28 -> 64ch tokens @ 56x56) + 6 blocks_c
  run_pe(W_X, 784, 28, 128, fea4, 64, 3, 2, 1, pe_c, true, false);
  for (int i = 0; i < 6; ++i) run_block(bc[i], 64, 56, 56, 4, 2, rel_c);
  k_head<<<12544, 32, 0, stream>>>(W_X, pre4_w, pre4_b, out_m4, 64);

  // Stage 4: pe_b (64ch 56x56 -> 16ch tokens @ 224x224), no fuse
  run_pe(W_X, 3136, 56, 64, nullptr, 16, 7, 4, 2, pe_b, false, true);
  k_head<<<200704, 32, 0, stream>>>(W_H, pre1_w, pre1_b, out_m1, 16);
}
